// Generator_45157286150811
// MI455X (gfx1250) — compile-verified, matrix operand order checked by reading, not run on hardware
//
#include <hip/hip_runtime.h>

// ---------------------------------------------------------------------------
// Swin shifted-window attention, fully fused, one block per (batch, window).
// f16 WMMA (v_wmma_f32_16x16x32_f16) for QKV GEMM, QK^T, PV, and out-proj.
// x is staged via CDNA5 GLOBAL_LOAD_ASYNC_TO_LDS_B128 (ASYNCcnt path);
// weights are pre-converted to f16 in d_ws by a prep kernel (L2-resident).
// ---------------------------------------------------------------------------

typedef __attribute__((ext_vector_type(16))) _Float16 v16h;
typedef __attribute__((ext_vector_type(8)))  _Float16 v8h;
typedef __attribute__((ext_vector_type(8)))  float    v8f;

#define WSZ   8      // window size
#define HD    32     // head dim
#define NH    8      // heads
#define CCH   256    // channels
#define P2    64     // tokens per window
#define HWIMG 256    // image H=W
#define NWX   32     // windows per row/col

#define WQKV_ELEMS (3 * CCH * CCH)     // 196608
#define WOUT_ELEMS (CCH * CCH)         // 65536
#define WF16_ELEMS (WQKV_ELEMS + WOUT_ELEMS)

#define SXS 264      // padded half-stride for 64x256 staging (264*2B = 132 banks)
#define SPS 72       // padded half-stride for probs scratch rows

// LDS carve (bytes)
#define OFF_A   0                       // 64*264*2 = 33792 (X f16, reused as attn-out)
#define OFF_Q   33792                   // also overlaid by 64KB f32 async stage
#define OFF_K   67584
#define OFF_V   101376
#define OFF_P   135168                  // 8 waves * 16*72*2 = 18432
#define OFF_REL 153600                  // 8*15*15*4 = 7200
#define SMEM_BYTES 160800

__device__ __forceinline__ v8f vzero() {
  v8f z;
#pragma unroll
  for (int i = 0; i < 8; ++i) z[i] = 0.0f;
  return z;
}

// Row-stationary fragment (A rows, or B columns of a row-major matrix):
// lane L holds row (row0 + L%16); k = col0 + (e/8)*16 + (L/16)*8 + (e%8).
__device__ __forceinline__ v16h frag_ld_lds_rm(const _Float16* base, int stride,
                                               int row0, int col0) {
  const int lane = threadIdx.x & 31;
  const _Float16* p = base + (row0 + (lane & 15)) * stride + col0 + ((lane >> 4) << 3);
  v16h f;
#pragma unroll
  for (int e = 0; e < 8; ++e) f[e] = p[e];
#pragma unroll
  for (int e = 0; e < 8; ++e) f[8 + e] = p[16 + e];
  return f;
}

// Column-stationary fragment (B columns of a row-major K x N buffer).
__device__ __forceinline__ v16h frag_ld_lds_cm(const _Float16* base, int stride,
                                               int row0, int col0) {
  const int lane = threadIdx.x & 31;
  const int c = col0 + (lane & 15);
  const int g = (lane >> 4) << 3;
  v16h f;
#pragma unroll
  for (int e = 0; e < 16; ++e) {
    int rr = row0 + ((e >> 3) << 4) + g + (e & 7);
    f[e] = base[rr * stride + c];
  }
  return f;
}

// B fragment from row-major f32 weights in global (fallback path).
__device__ __forceinline__ v16h frag_ld_gw32(const float* __restrict__ base, int stride,
                                             int row0, int col0) {
  const int lane = threadIdx.x & 31;
  const float* p = base + (size_t)(row0 + (lane & 15)) * stride + col0 + ((lane >> 4) << 3);
  v16h f;
#pragma unroll
  for (int e = 0; e < 8; ++e) f[e] = (_Float16)p[e];
#pragma unroll
  for (int e = 0; e < 8; ++e) f[8 + e] = (_Float16)p[16 + e];
  return f;
}

// B fragment from pre-converted row-major f16 weights: two 16B vector loads.
__device__ __forceinline__ v16h frag_ld_gw16(const _Float16* __restrict__ base, int stride,
                                             int row0, int col0) {
  const int lane = threadIdx.x & 31;
  const _Float16* p = base + (size_t)(row0 + (lane & 15)) * stride + col0 + ((lane >> 4) << 3);
  v8h lo = *(const v8h*)p;
  v8h hi = *(const v8h*)(p + 16);
  v16h f;
#pragma unroll
  for (int e = 0; e < 8; ++e) { f[e] = lo[e]; f[8 + e] = hi[e]; }
  return f;
}

template <bool WF16>
__device__ __forceinline__ v16h frag_ld_w(const void* base, int stride, int row0, int col0) {
  if (WF16) return frag_ld_gw16((const _Float16*)base, stride, row0, col0);
  else      return frag_ld_gw32((const float*)base, stride, row0, col0);
}

#define WMMA_F16(A, B, C) \
  __builtin_amdgcn_wmma_f32_16x16x32_f16(false, (A), false, (B), (short)0, (C), false, false)

// ---------------------------------------------------------------------------
// Prep kernel: convert w_qkv (768x256) and w_out (256x256) to f16 in d_ws.
// ---------------------------------------------------------------------------
__global__ void __launch_bounds__(256)
convert_weights_f16(const float* __restrict__ wq, const float* __restrict__ wo,
                    _Float16* __restrict__ dst) {
  int i = blockIdx.x * 256 + threadIdx.x;
  if (i < WQKV_ELEMS)      dst[i] = (_Float16)wq[i];
  else if (i < WF16_ELEMS) dst[i] = (_Float16)wo[i - WQKV_ELEMS];
}

// ---------------------------------------------------------------------------
// Fused window-attention kernel.
// ---------------------------------------------------------------------------
template <bool WF16>
__global__ void __launch_bounds__(256)
swin_window_attn(const float* __restrict__ x, const void* __restrict__ w_qkv,
                 const float* __restrict__ b_qkv, const float* __restrict__ rel_pos,
                 const void* __restrict__ w_out, const float* __restrict__ b_out,
                 float* __restrict__ out) {
  extern __shared__ char smem[];
  _Float16* sA = (_Float16*)(smem + OFF_A);   // X f16, later attn-out stage
  _Float16* sQ = (_Float16*)(smem + OFF_Q);
  _Float16* sK = (_Float16*)(smem + OFF_K);
  _Float16* sV = (_Float16*)(smem + OFF_V);
  _Float16* sP = (_Float16*)(smem + OFF_P);
  float*  sRel = (float*)(smem + OFF_REL);
  float* sStage = (float*)(smem + OFF_Q);     // 64KB raw f32 stage (overlay)

  const int bw = blockIdx.x;
  const int b  = bw >> 10;          // batch
  const int w  = bw & 1023;         // window id
  const int wy = w >> 5, wx = w & 31;
  const int tid  = threadIdx.x;
  const int wid  = tid >> 5;        // wave id == head id
  const int lane = tid & 31;
  const int nl   = lane & 15;       // N position within a 16-wide tile
  const int g    = lane >> 4;       // lane half (row offset 0/8 in C/D tiles)

  // ---- async-stage shifted window of x (f32) directly into LDS ------------
  {
    const int row = tid >> 2;          // 0..63 : window token
    const int cg  = tid & 3;           // chunk group within row
    const int pi = row >> 3, pj = row & 7;
    const int gh = (wy * WSZ + pi + 4) & 255;   // roll(-4) gather
    const int gw = (wx * WSZ + pj + 4) & 255;
    const float* rowbase = &x[(((size_t)b * HWIMG + gh) * HWIMG + gw) * CCH];
#pragma unroll
    for (int c = 0; c < 16; ++c) {
      const int chunk = cg + c * 4;    // 0..63, 16 bytes each
      // low 32 bits of a flat shared pointer == LDS byte address (aperture rule)
      unsigned lds = (unsigned)(size_t)(const void*)&sStage[row * CCH + chunk * 4];
      const float* ga = rowbase + chunk * 4;
      asm volatile("global_load_async_to_lds_b128 %0, %1, off"
                   :: "v"(lds), "v"(ga) : "memory");
    }
    asm volatile("s_wait_asynccnt 0x0" ::: "memory");
  }
  // ---- stage rel_pos bias table -------------------------------------------
  for (int i = tid; i < NH * 15 * 15; i += 256) sRel[i] = rel_pos[i];
  __syncthreads();

  // ---- convert staged f32 window -> f16 A-operand buffer ------------------
  {
    const int col = tid;  // 0..255 channels
#pragma unroll 4
    for (int row = 0; row < P2; ++row)
      sA[row * SXS + col] = (_Float16)sStage[row * CCH + col];
  }
  __syncthreads();

  // ---- QKV projection: (64x256) @ (256x768)^T, WMMA 16x16x32 --------------
  const float qscale = 0.17677669529663687f;  // HD^-0.5
  for (int nti = 0; nti < 6; ++nti) {
    const int nt = wid * 6 + nti;
    const int j0 = nt << 4;                   // output column base (0..752)
    v8f acc[4];
#pragma unroll
    for (int mt = 0; mt < 4; ++mt) acc[mt] = vzero();
    for (int ks = 0; ks < 8; ++ks) {
      v16h bf = frag_ld_w<WF16>(w_qkv, CCH, j0, ks * 32);
#pragma unroll
      for (int mt = 0; mt < 4; ++mt) {
        v16h af = frag_ld_lds_rm(sA, SXS, mt << 4, ks * 32);
        acc[mt] = WMMA_F16(af, bf, acc[mt]);
      }
    }
    // epilogue: bias, q-scale, scatter to sQ/sK/sV as f16
    const int j = j0 + nl;
    const float bias = b_qkv[j];
    _Float16* dst;
    int jj;
    float scale;
    if (j0 < 256)      { dst = sQ; jj = j;        scale = qscale; }
    else if (j0 < 512) { dst = sK; jj = j - 256;  scale = 1.0f;   }
    else               { dst = sV; jj = j - 512;  scale = 1.0f;   }
#pragma unroll
    for (int mt = 0; mt < 4; ++mt)
#pragma unroll
      for (int r = 0; r < 8; ++r) {
        int row = (mt << 4) + r + (g << 3);
        dst[row * SXS + jj] = (_Float16)((acc[mt][r] + bias) * scale);
      }
  }
  __syncthreads();

  // ---- per-head attention: wave `wid` handles head h = wid ---------------
  const int h = wid;
  v16h kf[4], vf[4];
#pragma unroll
  for (int t = 0; t < 4; ++t)                        // B = K^T, cols = K rows
    kf[t] = frag_ld_lds_rm(sK, SXS, t << 4, h * HD);
#pragma unroll
  for (int ks = 0; ks < 2; ++ks)                     // B = V (64x32), k = token
#pragma unroll
    for (int nc = 0; nc < 2; ++nc)
      vf[ks * 2 + nc] = frag_ld_lds_cm(sV, SXS, ks * 32, h * HD + nc * 16);

  _Float16* sPw = sP + wid * 16 * SPS;               // per-wave probs scratch
  const float* relh = sRel + h * 225;
  const bool lastRow = (wy == NWX - 1), lastCol = (wx == NWX - 1);

  for (int mt = 0; mt < 4; ++mt) {
    v8f sim[4];
#pragma unroll
    for (int t = 0; t < 4; ++t) sim[t] = vzero();
    v16h qf = frag_ld_lds_rm(sQ, SXS, mt << 4, h * HD);
#pragma unroll
    for (int t = 0; t < 4; ++t) sim[t] = WMMA_F16(qf, kf[t], sim[t]);

    // relative-position bias + shifted-window mask (C/D layout coordinates)
#pragma unroll
    for (int t = 0; t < 4; ++t) {
      const int q0 = (t << 4) + nl;                  // key token
      const int qi = q0 >> 3, qj = q0 & 7;
#pragma unroll
      for (int r = 0; r < 8; ++r) {
        const int p = (mt << 4) + r + (g << 3);      // query token
        const int pi = p >> 3, pj = p & 7;
        float s = sim[t][r] + relh[(pi - qi + 7) * 15 + (pj - qj + 7)];
        bool masked = (lastRow && ((pi < 4) != (qi < 4))) ||
                      (lastCol && ((pj < 4) != (qj < 4)));
        sim[t][r] = masked ? -1e30f : s;
      }
    }
    // softmax over the 64-wide row: reduce 4 tiles then across 16 lanes
#pragma unroll
    for (int r = 0; r < 8; ++r) {
      float m = fmaxf(fmaxf(sim[0][r], sim[1][r]), fmaxf(sim[2][r], sim[3][r]));
#pragma unroll
      for (int off = 8; off >= 1; off >>= 1) m = fmaxf(m, __shfl_xor(m, off, 16));
      float sum = 0.0f;
#pragma unroll
      for (int t = 0; t < 4; ++t) {
        float e = __expf(sim[t][r] - m);
        sim[t][r] = e;
        sum += e;
      }
#pragma unroll
      for (int off = 8; off >= 1; off >>= 1) sum += __shfl_xor(sum, off, 16);
      float inv = __builtin_amdgcn_rcpf(sum);
#pragma unroll
      for (int t = 0; t < 4; ++t) sim[t][r] *= inv;
    }
    // D-layout -> A-layout relayout for probs via per-wave LDS bounce
#pragma unroll
    for (int t = 0; t < 4; ++t)
#pragma unroll
      for (int r = 0; r < 8; ++r)
        sPw[(r + (g << 3)) * SPS + (t << 4) + nl] = (_Float16)sim[t][r];
    __syncthreads();

    v16h pf0 = frag_ld_lds_rm(sPw, SPS, 0, 0);
    v16h pf1 = frag_ld_lds_rm(sPw, SPS, 0, 32);
    v8f oacc[2];
#pragma unroll
    for (int nc = 0; nc < 2; ++nc) {
      oacc[nc] = vzero();
      oacc[nc] = WMMA_F16(pf0, vf[nc],     oacc[nc]);
      oacc[nc] = WMMA_F16(pf1, vf[2 + nc], oacc[nc]);
    }
    // stash attention output (f16) into sA at this head's channel slice
#pragma unroll
    for (int nc = 0; nc < 2; ++nc)
#pragma unroll
      for (int r = 0; r < 8; ++r)
        sA[((mt << 4) + r + (g << 3)) * SXS + h * HD + nc * 16 + nl] =
            (_Float16)oacc[nc][r];
    __syncthreads();
  }

  // ---- output projection: (64x256) @ (256x256)^T, then rolled NT scatter --
  const size_t obase = (size_t)b * HWIMG * HWIMG * CCH;
  for (int nti = 0; nti < 2; ++nti) {
    const int nt = wid * 2 + nti;
    const int j0 = nt << 4;
    v8f acc[4];
#pragma unroll
    for (int mt = 0; mt < 4; ++mt) acc[mt] = vzero();
    for (int ks = 0; ks < 8; ++ks) {
      v16h bf = frag_ld_w<WF16>(w_out, CCH, j0, ks * 32);
#pragma unroll
      for (int mt = 0; mt < 4; ++mt) {
        v16h af = frag_ld_lds_rm(sA, SXS, mt << 4, ks * 32);
        acc[mt] = WMMA_F16(af, bf, acc[mt]);
      }
    }
    const int j = j0 + nl;
    const float bias = b_out[j];
#pragma unroll
    for (int mt = 0; mt < 4; ++mt)
#pragma unroll
      for (int r = 0; r < 8; ++r) {
        const int p = (mt << 4) + r + (g << 3);
        const int pi = p >> 3, pj = p & 7;
        const int gh = (wy * WSZ + pi + 4) & 255;   // roll(+4) scatter
        const int gw = (wx * WSZ + pj + 4) & 255;
        __builtin_nontemporal_store(
            acc[mt][r] + bias,
            &out[obase + (((size_t)gh * HWIMG + gw) * CCH) + j]);
      }
  }
}

extern "C" void kernel_launch(void* const* d_in, const int* in_sizes, int n_in,
                              void* d_out, int out_size, void* d_ws, size_t ws_size,
                              hipStream_t stream) {
  const float* x       = (const float*)d_in[0];
  const float* w_qkv   = (const float*)d_in[1];
  const float* b_qkv   = (const float*)d_in[2];
  const float* rel_pos = (const float*)d_in[3];
  const float* w_out   = (const float*)d_in[4];
  const float* b_out   = (const float*)d_in[5];
  float* out           = (float*)d_out;

  const int B = in_sizes[0] / (HWIMG * HWIMG * CCH);   // = 2
  const int blocks = B * NWX * NWX;                    // one block per window

  const bool wf16 = (ws_size >= (size_t)WF16_ELEMS * sizeof(_Float16));

  if (wf16) {
    _Float16* wfp = (_Float16*)d_ws;
    convert_weights_f16<<<dim3((WF16_ELEMS + 255) / 256), dim3(256), 0, stream>>>(
        w_qkv, w_out, wfp);
    (void)hipFuncSetAttribute((const void*)swin_window_attn<true>,
                              hipFuncAttributeMaxDynamicSharedMemorySize, SMEM_BYTES);
    swin_window_attn<true><<<dim3(blocks), dim3(256), SMEM_BYTES, stream>>>(
        x, (const void*)wfp, b_qkv, rel_pos, (const void*)(wfp + WQKV_ELEMS), b_out, out);
  } else {
    (void)hipFuncSetAttribute((const void*)swin_window_attn<false>,
                              hipFuncAttributeMaxDynamicSharedMemorySize, SMEM_BYTES);
    swin_window_attn<false><<<dim3(blocks), dim3(256), SMEM_BYTES, stream>>>(
        x, (const void*)w_qkv, b_qkv, rel_pos, (const void*)w_out, b_out, out);
  }
}
